// Loss_52802327937685
// MI455X (gfx1250) — matrix operand order repaired and checked by reading, not compiled, not run on hardware
//
#include <hip/hip_runtime.h>
#include <math.h>
#include <stdint.h>

// ---------------- CDNA5 types ----------------
typedef __bf16 v16bf __attribute__((ext_vector_type(16)));
typedef float  v8f   __attribute__((ext_vector_type(8)));
typedef unsigned int u32v4 __attribute__((ext_vector_type(4)));
typedef unsigned int u32x4 __attribute__((ext_vector_type(4)));
typedef int i32x4 __attribute__((ext_vector_type(4)));
typedef int i32x8 __attribute__((ext_vector_type(8)));
typedef union { v16bf v; unsigned short u[16]; u32v4 q[2]; } frag16;

// LDS tile geometry: 64 rows x 32 halfwords, padded to 40 halfwords (80B) per row
#define LDS_STRIDE 40
#define TILE_HW (64 * LDS_STRIDE)

// ---------------- TDM (Tensor Data Mover) ----------------
#if defined(__AMDGCN__) && __has_builtin(__builtin_amdgcn_tensor_load_to_lds)
#define HAS_TDM 1
#if __has_include(<hip/amd_detail/amd_gfx1250_TDM.h>)
#define TDM_CALL(g0, g1, z4a, z4b, z8) \
    __builtin_amdgcn_tensor_load_to_lds((g0), (g1), (z4a), (z4b), (z8), 0)
#else
#define TDM_CALL(g0, g1, z4a, z4b, z8) \
    __builtin_amdgcn_tensor_load_to_lds((g0), (g1), (z4a), (z4b), 0)
#endif

#define RFLU(x) ((unsigned)__builtin_amdgcn_readfirstlane((int)(x)))

// Load a 2D tile (rows x 32 bf16 elems) from a row-major tensor into LDS with
// 80B padded rows.  D# layout per CDNA5 ISA ch.8:
//  group0: [1:0]=count=1, [63:32]=lds_addr, [120:64]=global_addr, [127:126]=type=2
//  group1: [17:16]=data_size=1(2B), [20]=pad_en, [24:22]=pad_interval(3 -> 16 dw),
//          [31:25]=pad_amount(3 -> 4 dw), tensor_dim0/1, tile 32 x rows,
//          tensor_dim0_stride
__device__ __forceinline__ void tdm_load_tile(const void* gptr, const void* ldsptr,
                                              unsigned rows, unsigned tdim0,
                                              unsigned tdim1,
                                              unsigned long long stride0) {
    unsigned long long ga = (unsigned long long)(uintptr_t)gptr;
    unsigned la = (unsigned)(uintptr_t)ldsptr;  // low 32 bits == LDS byte address
    u32x4 g0 = {RFLU(1u), RFLU(la), RFLU((unsigned)ga),
                RFLU(((unsigned)((ga >> 32) & 0x01FFFFFFull)) | 0x80000000u)};
    unsigned w0 = 0x00010000u | (1u << 20) | (3u << 22) | (3u << 25);
    i32x8 g1 = {(int)RFLU(w0),
                (int)RFLU((tdim0 & 0xFFFFu) << 16),
                (int)RFLU((tdim0 >> 16) | ((tdim1 & 0xFFFFu) << 16)),
                (int)RFLU((tdim1 >> 16) | (32u << 16)),          // tile_dim0 = 32
                (int)RFLU(rows),                                  // tile_dim1
                (int)RFLU((unsigned)(stride0 & 0xFFFFFFFFull)),
                (int)RFLU((unsigned)((stride0 >> 32) & 0xFFFFull)),
                0};
    i32x4 z4 = {0, 0, 0, 0};
    i32x8 z8 = {0, 0, 0, 0, 0, 0, 0, 0};
    TDM_CALL(g0, g1, z4, z4, z8);
}
#endif

// ---------------- scalar helpers ----------------
__device__ __forceinline__ float bf2f(unsigned short h) {
    return __uint_as_float(((unsigned int)h) << 16);
}
__device__ __forceinline__ unsigned short f2bf(float f) {
    unsigned int u = __float_as_uint(f);
    return (unsigned short)((u + 0x7FFFu + ((u >> 16) & 1u)) >> 16);  // RNE
}
__device__ __forceinline__ unsigned int f2ord(float f) {
    unsigned int u = __float_as_uint(f);
    return (u & 0x80000000u) ? ~u : (u | 0x80000000u);
}
__device__ __forceinline__ float ord2f(unsigned int e) {
    return __uint_as_float((e & 0x80000000u) ? (e & 0x7FFFFFFFu) : ~e);
}
__device__ __forceinline__ float h16_min(float v) {
    for (int m = 1; m < 16; m <<= 1) v = fminf(v, __shfl_xor(v, m, 32));
    return v;
}
__device__ __forceinline__ float h16_max(float v) {
    for (int m = 1; m < 16; m <<= 1) v = fmaxf(v, __shfl_xor(v, m, 32));
    return v;
}
__device__ __forceinline__ float h16_sum(float v) {
    for (int m = 1; m < 16; m <<= 1) v += __shfl_xor(v, m, 32);
    return v;
}
__device__ __forceinline__ float blockSum256(float v) {
    __shared__ float sm[8];
    for (int m = 16; m > 0; m >>= 1) v += __shfl_xor(v, m, 32);
    int lane = threadIdx.x & 31, wv = threadIdx.x >> 5;
    if (lane == 0) sm[wv] = v;
    __syncthreads();
    float r = 0.f;
    if (threadIdx.x < 8) r = sm[threadIdx.x];
    if (wv == 0) for (int m = 4; m > 0; m >>= 1) r += __shfl_xor(r, m, 32);
    __syncthreads();
    return r;
}

// Per-lane fragment fetch from padded LDS tiles (two ds_load_b128 per operand).
// A tile [m][k]: lane<16 -> k 0..7 & 16..23 ; lane>=16 -> 8..15 & 24..31
// B tile [n][k]: lane<16 -> k 0..7 & 16..23 ; lane>=16 -> 8..15 & 24..31
__device__ __forceinline__ frag16 fragA(const u32v4* t4, int row, int hi) {
    frag16 f;
    f.q[0] = t4[row * (LDS_STRIDE / 8) + hi];
    f.q[1] = t4[row * (LDS_STRIDE / 8) + 2 + hi];
    return f;
}
__device__ __forceinline__ frag16 fragB(const u32v4* t4, int row, int hi) {
    frag16 f;
    f.q[0] = t4[row * (LDS_STRIDE / 8) + hi];
    f.q[1] = t4[row * (LDS_STRIDE / 8) + 2 + hi];
    return f;
}

// ---------------- small utility kernels ----------------
__global__ void fill_u32(unsigned int* p, unsigned int val, size_t n) {
    size_t stride = (size_t)gridDim.x * blockDim.x;
    for (size_t i = (size_t)blockIdx.x * blockDim.x + threadIdx.x; i < n; i += stride)
        p[i] = val;
}

__global__ void f32_to_bf16_k(const float* __restrict__ in,
                              unsigned short* __restrict__ out, size_t n) {
    size_t stride = (size_t)gridDim.x * blockDim.x;
    for (size_t i = (size_t)blockIdx.x * blockDim.x + threadIdx.x; i < n; i += stride)
        out[i] = f2bf(in[i]);
}

__global__ void fold_bn(const float* __restrict__ W, const float* __restrict__ b,
                        const float* __restrict__ g, const float* __restrict__ be,
                        const float* __restrict__ mn, const float* __restrict__ vr,
                        unsigned short* __restrict__ Wb, float* __restrict__ bout,
                        int Cout, int Kc) {
    size_t i = (size_t)blockIdx.x * blockDim.x + threadIdx.x;
    size_t total = (size_t)Cout * Kc;
    if (i < total) {
        int co = (int)(i / Kc);
        float inv = g[co] * rsqrtf(vr[co] + 1e-5f);
        Wb[i] = f2bf(W[i] * inv);
    }
    if (i < (size_t)Cout) {
        float inv = g[i] * rsqrtf(vr[i] + 1e-5f);
        bout[i] = b[i] * inv + be[i] - mn[i] * inv;
    }
}

__global__ void pool2x2(const unsigned short* __restrict__ in,
                        unsigned short* __restrict__ out, int C, int H, int Wd) {
    int HO = H >> 1, WO = Wd >> 1;
    size_t total = (size_t)C * HO * WO;
    size_t stride = (size_t)gridDim.x * blockDim.x;
    for (size_t i = (size_t)blockIdx.x * blockDim.x + threadIdx.x; i < total; i += stride) {
        int wo = (int)(i % WO);
        size_t t = i / WO;
        int ho = (int)(t % HO);
        int c  = (int)(t / HO);
        const unsigned short* base = in + ((size_t)c * H + 2 * ho) * Wd + 2 * wo;
        float a = bf2f(base[0]), b2 = bf2f(base[1]);
        float d = bf2f(base[Wd]), e = bf2f(base[Wd + 1]);
        out[i] = f2bf(fmaxf(fmaxf(a, b2), fmaxf(d, e)));
    }
}

// ---------------- implicit-GEMM conv via WMMA bf16 ----------------
// Block tile: 64 out-channels x 64 pixels, K stepped by 32.
// 8 waves = 2(M) x 4(N); each wave: 2 accumulators (two 16-row M subtiles).
// Weights streamed by TDM with double-buffered LDS (DMA overlaps compute+im2col).
__global__ __launch_bounds__(256)
void conv_wmma_bf16(const unsigned short* __restrict__ in,
                    const unsigned short* __restrict__ wgt,
                    const float* __restrict__ bias,
                    unsigned short* __restrict__ out,
                    int Cin, int Cout, int H, int Wd) {
    const int K      = Cin * 9;
    const int NPIX   = H * Wd;
    const int n0     = blockIdx.x * 64;
    const int m0     = blockIdx.y * 64;
    const int ksteps = (K + 31) >> 5;

    __shared__ __align__(16) unsigned short shA[2 * TILE_HW];  // ping-pong weights
    __shared__ __align__(16) unsigned short shB[TILE_HW];      // im2col^T [n][k]

    const int tid  = threadIdx.x;
    const int lane = tid & 31;
    const int wave = tid >> 5;
    const int wm   = wave >> 2;   // 0..1
    const int wn   = wave & 3;    // 0..3
    const int hi   = (lane < 16) ? 0 : 1;

    // per-thread im2col coordinates: one pixel, 8 contiguous k values
    const int nB  = tid >> 2;          // 0..63
    const int kb  = (tid & 3) << 3;    // 0,8,16,24
    const int pB  = n0 + nB;
    const int pyB = pB / Wd;
    const int pxB = pB - pyB * Wd;

#ifdef HAS_TDM
    if (tid == 0)
        tdm_load_tile(wgt + (size_t)m0 * K, shA, 64,
                      (unsigned)K, (unsigned)(Cout - m0), (unsigned long long)K);
#endif

    v8f acc0 = {}, acc1 = {};
    for (int it = 0; it < ksteps; ++it) {
        const int k0 = it << 5;

        // cooperative im2col fill (one aligned 16B LDS store per thread)
        if (it + 1 < ksteps)
            __builtin_prefetch(&in[((size_t)((k0 + 32) / 9) * H + pyB) * Wd + pxB], 0, 0);
        union { unsigned short u[8]; u32v4 q; } t;
#pragma unroll
        for (int j = 0; j < 8; j++) {
            int gk = k0 + kb + j;
            unsigned short v = 0;
            if (gk < K) {
                int ci = gk / 9, rs = gk - ci * 9;
                int r = rs / 3, s = rs - r * 3;
                int y = pyB + r - 1, x = pxB + s - 1;
                if ((unsigned)y < (unsigned)H && (unsigned)x < (unsigned)Wd)
                    v = in[((size_t)ci * H + y) * Wd + x];
            }
            t.u[j] = v;
        }
        *(u32v4*)&shB[nB * LDS_STRIDE + kb] = t.q;

#ifdef HAS_TDM
        if (tid == 0) {   // pipeline: issue next tile, then wait for current
            if (it + 1 < ksteps) {
                tdm_load_tile(wgt + (size_t)m0 * K + (k0 + 32),
                              shA + ((it + 1) & 1) * TILE_HW, 64,
                              (unsigned)(K - (k0 + 32)), (unsigned)(Cout - m0),
                              (unsigned long long)K);
                __builtin_amdgcn_s_wait_tensorcnt(1);
            } else {
                __builtin_amdgcn_s_wait_tensorcnt(0);
            }
        }
#else
        {
            unsigned short* A = shA + (it & 1) * TILE_HW;
            for (int i = tid; i < 64 * 32; i += 256) {
                int m = i >> 5, k = i & 31, gk = k0 + k;
                A[m * LDS_STRIDE + k] =
                    (gk < K) ? wgt[(size_t)(m0 + m) * K + gk] : (unsigned short)0;
            }
        }
#endif
        __syncthreads();

        const u32v4* A4 = (const u32v4*)(shA + (it & 1) * TILE_HW);
        const u32v4* B4 = (const u32v4*)shB;
        const int ml = wm * 32 + (lane & 15);
        const int nl = wn * 16 + (lane & 15);
        frag16 a0 = fragA(A4, ml, hi);
        frag16 a1 = fragA(A4, ml + 16, hi);
        frag16 b  = fragB(B4, nl, hi);
        acc0 = __builtin_amdgcn_wmma_f32_16x16x32_bf16(false, a0.v, false, b.v,
                                                       (short)0, acc0, false, false);
        acc1 = __builtin_amdgcn_wmma_f32_16x16x32_bf16(false, a1.v, false, b.v,
                                                       (short)0, acc1, false, false);
        __syncthreads();
    }

#pragma unroll
    for (int vv = 0; vv < 8; vv++) {
        int md = (lane < 16) ? vv : vv + 8;
        int gp = n0 + wn * 16 + (lane & 15);
        int gm = m0 + wm * 32 + md;
        out[(size_t)gm * NPIX + gp] = f2bf(fmaxf(acc0[vv] + bias[gm], 0.f));
        int gm1 = gm + 16;
        out[(size_t)gm1 * NPIX + gp] = f2bf(fmaxf(acc1[vv] + bias[gm1], 0.f));
    }
}

// ---------------- contextual loss ----------------
__global__ __launch_bounds__(256)
void ctx_normalize(const unsigned short* __restrict__ f,
                   unsigned short* __restrict__ outn, int C, int P) {
    int c = blockIdx.x;
    const unsigned short* fc = f + (size_t)c * P;
    float s = 0.f, ss = 0.f;
    for (int p = threadIdx.x; p < P; p += blockDim.x) {
        float x = bf2f(fc[p]);
        s += x; ss += x * x;
    }
    float ts  = blockSum256(s);
    float tss = blockSum256(ss);
    __shared__ float mb[2];
    if (threadIdx.x == 0) { mb[0] = ts / (float)P; mb[1] = sqrtf(tss) + 1e-5f; }
    __syncthreads();
    float mean = mb[0], denom = mb[1];
    for (int p = threadIdx.x; p < P; p += blockDim.x) {
        float x = (bf2f(fc[p]) - mean) / denom;
        outn[(size_t)p * C + c] = f2bf(x);
    }
}

// D = Xn @ Yn^T, 64x64 block tile; both operand tiles TDM-loaded, double buffered.
// pass 0: per-row min of D; pass 1: per-row Z = sum exp(s), smax = max s.
__global__ __launch_bounds__(256)
void ctx_gemm_pass(const unsigned short* __restrict__ Xn,
                   const unsigned short* __restrict__ Yn,
                   int C, int P,
                   unsigned int* __restrict__ rowMinU,
                   float* __restrict__ rowZ,
                   unsigned int* __restrict__ rowSmaxU, int pass) {
    const int n0 = blockIdx.x * 64;
    const int m0 = blockIdx.y * 64;
    const int ksteps = C >> 5;   // C in {128,256,512}
    __shared__ __align__(16) unsigned short shA[2 * TILE_HW];  // [m][k] ping-pong
    __shared__ __align__(16) unsigned short shB[2 * TILE_HW];  // [n][k] ping-pong
    const int tid  = threadIdx.x;
    const int lane = tid & 31;
    const int wave = tid >> 5;
    const int wm   = wave >> 2;
    const int wn   = wave & 3;
    const int hi   = (lane < 16) ? 0 : 1;

#ifdef HAS_TDM
    if (tid == 0) {
        tdm_load_tile(Xn + (size_t)m0 * C, shA, 64,
                      (unsigned)C, (unsigned)(P - m0), (unsigned long long)C);
        tdm_load_tile(Yn + (size_t)n0 * C, shB, 64,
                      (unsigned)C, (unsigned)(P - n0), (unsigned long long)C);
    }
#endif

    v8f acc0 = {}, acc1 = {};
    for (int it = 0; it < ksteps; ++it) {
        const int k0 = it << 5;
#ifdef HAS_TDM
        if (tid == 0) {   // pipeline: issue next pair, wait for current pair
            if (it + 1 < ksteps) {
                tdm_load_tile(Xn + (size_t)m0 * C + (k0 + 32),
                              shA + ((it + 1) & 1) * TILE_HW, 64,
                              (unsigned)(C - (k0 + 32)), (unsigned)(P - m0),
                              (unsigned long long)C);
                tdm_load_tile(Yn + (size_t)n0 * C + (k0 + 32),
                              shB + ((it + 1) & 1) * TILE_HW, 64,
                              (unsigned)(C - (k0 + 32)), (unsigned)(P - n0),
                              (unsigned long long)C);
                __builtin_amdgcn_s_wait_tensorcnt(2);
            } else {
                __builtin_amdgcn_s_wait_tensorcnt(0);
            }
        }
#else
        {
            unsigned short* A = shA + (it & 1) * TILE_HW;
            unsigned short* B = shB + (it & 1) * TILE_HW;
            for (int i = tid; i < 64 * 32; i += 256) {
                int m = i >> 5, k = i & 31;
                A[m * LDS_STRIDE + k] = Xn[(size_t)(m0 + m) * C + k0 + k];
            }
            for (int i = tid; i < 64 * 32; i += 256) {
                int n = i >> 5, k = i & 31;
                B[n * LDS_STRIDE + k] = Yn[(size_t)(n0 + n) * C + k0 + k];
            }
        }
#endif
        __syncthreads();
        const u32v4* A4 = (const u32v4*)(shA + (it & 1) * TILE_HW);
        const u32v4* B4 = (const u32v4*)(shB + (it & 1) * TILE_HW);
        const int ml = wm * 32 + (lane & 15);
        const int nl = wn * 16 + (lane & 15);
        frag16 a0 = fragA(A4, ml, hi);
        frag16 a1 = fragA(A4, ml + 16, hi);
        frag16 b  = fragB(B4, nl, hi);
        acc0 = __builtin_amdgcn_wmma_f32_16x16x32_bf16(false, a0.v, false, b.v,
                                                       (short)0, acc0, false, false);
        acc1 = __builtin_amdgcn_wmma_f32_16x16x32_bf16(false, a1.v, false, b.v,
                                                       (short)0, acc1, false, false);
        __syncthreads();
    }

    if (pass == 0) {
#pragma unroll
        for (int vv = 0; vv < 8; vv++) {
            float mn0 = h16_min(acc0[vv]);
            float mn1 = h16_min(acc1[vv]);
            if ((lane & 15) == 0) {
                int gm = m0 + wm * 32 + ((lane < 16) ? vv : vv + 8);
                atomicMin(&rowMinU[gm], f2ord(mn0));
                atomicMin(&rowMinU[gm + 16], f2ord(mn1));
            }
        }
    } else {
#pragma unroll
        for (int vv = 0; vv < 8; vv++) {
            int gm = m0 + wm * 32 + ((lane < 16) ? vv : vv + 8);
#pragma unroll
            for (int sub = 0; sub < 2; sub++) {
                float d = (sub == 0) ? acc0[vv] : acc1[vv];
                int row = gm + sub * 16;
                float dmin = ord2f(rowMinU[row]);
                float s = (1.f - d / (dmin + 1e-5f)) * 10.f;  // / BANDWIDTH(0.1)
                float e = __expf(s);
                float es = h16_sum(e);
                float sm = h16_max(s);
                if ((lane & 15) == 0) {
                    atomicAdd(&rowZ[row], es);
                    atomicMax(&rowSmaxU[row], f2ord(sm));
                }
            }
        }
    }
}

__global__ __launch_bounds__(256)
void ctx_row_accum(const float* __restrict__ Z, const unsigned int* __restrict__ smaxU,
                   float* __restrict__ inside, int P) {
    float s = 0.f;
    int stride = gridDim.x * blockDim.x;
    for (int i = blockIdx.x * blockDim.x + threadIdx.x; i < P; i += stride)
        s += __expf(ord2f(smaxU[i])) / Z[i];
    s = blockSum256(s);
    if (threadIdx.x == 0) atomicAdd(inside, s);
}

__global__ void ctx_finalize(const float* __restrict__ inside, float* __restrict__ out,
                             float wstage, float nch) {
    if (threadIdx.x == 0 && blockIdx.x == 0)
        atomicAdd(out, -0.2f * wstage * __logf(inside[0] / nch));  // CONTEXTUAL_LAMDA=0.2
}

// ---------------- simple losses ----------------
__global__ __launch_bounds__(256)
void sumsq_bf16(const unsigned short* __restrict__ a, const unsigned short* __restrict__ b,
                float* __restrict__ out, float scale, size_t n) {
    float s = 0.f;
    size_t stride = (size_t)gridDim.x * blockDim.x;
    for (size_t i = (size_t)blockIdx.x * blockDim.x + threadIdx.x; i < n; i += stride) {
        float d = bf2f(a[i]) - bf2f(b[i]);
        s += d * d;
    }
    s = blockSum256(s);
    if (threadIdx.x == 0) atomicAdd(out, scale * s);
}

__global__ __launch_bounds__(256)
void l1_f32(const float* __restrict__ a, const float* __restrict__ b,
            float* __restrict__ out, float scale, size_t off, size_t n) {
    float s = 0.f;
    size_t stride = (size_t)gridDim.x * blockDim.x;
    for (size_t i = (size_t)blockIdx.x * blockDim.x + threadIdx.x; i < n; i += stride)
        s += fabsf(a[off + i] - b[off + i]);
    s = blockSum256(s);
    if (threadIdx.x == 0) atomicAdd(out, scale * s);
}

// ---------------- host orchestration ----------------
extern "C" void kernel_launch(void* const* d_in, const int* in_sizes, int n_in,
                              void* d_out, int out_size, void* d_ws, size_t ws_size,
                              hipStream_t stream) {
    (void)in_sizes; (void)n_in; (void)out_size; (void)ws_size;

    static const int CIN_[14]  = {3,64,64,128,128,256,256,256,256,512,512,512,512,512};
    static const int COUT_[14] = {64,64,128,128,256,256,256,256,512,512,512,512,512,512};
    static const int CAPC[4] = {128, 256, 512, 512};
    static const int CAPP[4] = {9216, 2304, 576, 576};
    static const float WST[4] = {0.1f, 0.6f, 0.7f, 0.8f};

    size_t woff[14], boff[14], wtot = 0, btot = 0;
    for (int l = 0; l < 14; l++) {
        woff[l] = wtot; wtot += (size_t)COUT_[l] * CIN_[l] * 9;
        boff[l] = btot; btot += COUT_[l];
    }

    size_t off = 0;
    char* ws = (char*)d_ws;
    auto alloc = [&](size_t bytes) -> size_t {
        size_t o = off; off += (bytes + 255) & ~(size_t)255; return o;
    };
    unsigned short* wbf   = (unsigned short*)(ws + alloc(wtot * 2));
    float*          bfold = (float*)(ws + alloc(btot * 4));
    unsigned short* actA  = (unsigned short*)(ws + alloc((size_t)64 * 192 * 192 * 2));
    unsigned short* actB  = (unsigned short*)(ws + alloc((size_t)64 * 192 * 192 * 2));
    unsigned short* cap[3][4];
    for (int j = 0; j < 3; j++)
        for (int s = 0; s < 4; s++)
            cap[j][s] = (unsigned short*)(ws + alloc((size_t)CAPC[s] * CAPP[s] * 2));
    unsigned short* Xn = (unsigned short*)(ws + alloc((size_t)9216 * 128 * 2));
    unsigned short* Yn = (unsigned short*)(ws + alloc((size_t)9216 * 128 * 2));
    unsigned int* rowMinU  = (unsigned int*)(ws + alloc(9216 * 4));
    unsigned int* rowSmaxU = (unsigned int*)(ws + alloc(9216 * 4));
    float*        rowZ     = (float*)(ws + alloc(9216 * 4));
    float*        insideBuf = (float*)(ws + alloc(256));

    // 1) fold BN into bf16 weights
    for (int l = 0; l < 14; l++) {
        const float* Wc = (const float*)d_in[4 + 6 * l + 0];
        const float* bb = (const float*)d_in[4 + 6 * l + 1];
        const float* gg = (const float*)d_in[4 + 6 * l + 2];
        const float* be = (const float*)d_in[4 + 6 * l + 3];
        const float* mn = (const float*)d_in[4 + 6 * l + 4];
        const float* vr = (const float*)d_in[4 + 6 * l + 5];
        size_t tot = (size_t)COUT_[l] * CIN_[l] * 9;
        fold_bn<<<(int)((tot + 255) / 256), 256, 0, stream>>>(
            Wc, bb, gg, be, mn, vr, wbf + woff[l], bfold + boff[l], COUT_[l], CIN_[l] * 9);
    }

    // 2) zero the scalar output
    fill_u32<<<1, 64, 0, stream>>>((unsigned int*)d_out, 0u, (size_t)1);

    // 3) VGG trunk on res(0), gt(d_in[2]), ref(d_in[3])
    const float* imgs[3] = {(const float*)d_in[0], (const float*)d_in[2], (const float*)d_in[3]};
    for (int j = 0; j < 3; j++) {
        size_t nimg = (size_t)3 * 192 * 192;
        f32_to_bf16_k<<<(int)((nimg + 255) / 256), 256, 0, stream>>>(imgs[j], actA, nimg);
        unsigned short* cur = actA;
        int H = 192, Wd = 192, stage = 0;
        for (int l = 0; l < 14; l++) {
            bool isCap = (l == 3 || l == 5 || l == 9 || l == 13);
            unsigned short* outp = isCap ? cap[j][stage] : ((cur == actA) ? actB : actA);
            dim3 grid(H * Wd / 64, COUT_[l] / 64);
            conv_wmma_bf16<<<grid, 256, 0, stream>>>(cur, wbf + woff[l], bfold + boff[l],
                                                     outp, CIN_[l], COUT_[l], H, Wd);
            cur = outp;
            if (isCap) stage++;
            if (l == 1 || l == 3 || l == 7 || l == 11) {
                unsigned short* pout = (cur == actA) ? actB : actA;
                size_t tot = (size_t)COUT_[l] * (H / 2) * (Wd / 2);
                pool2x2<<<(int)((tot + 255) / 256), 256, 0, stream>>>(cur, pout, COUT_[l], H, Wd);
                cur = pout; H /= 2; Wd /= 2;
            }
        }
    }

    // 4) perceptual: ||relu5_2(res) - relu5_2(gt)||^2 * 0.001
    sumsq_bf16<<<64, 256, 0, stream>>>(cap[0][3], cap[1][3], (float*)d_out,
                                       0.001f, (size_t)512 * 576);
    // 5) L1 on channels 1..2 of res vs gt, * 2.0
    l1_f32<<<64, 256, 0, stream>>>((const float*)d_in[0], (const float*)d_in[2],
                                   (float*)d_out, 2.0f, (size_t)192 * 192, (size_t)2 * 192 * 192);

    // 6) contextual stages (res vs ref)
    for (int s = 0; s < 4; s++) {
        int C = CAPC[s], P = CAPP[s];
        ctx_normalize<<<C, 256, 0, stream>>>(cap[0][s], Xn, C, P);
        ctx_normalize<<<C, 256, 0, stream>>>(cap[2][s], Yn, C, P);
        fill_u32<<<(P + 255) / 256, 256, 0, stream>>>(rowMinU, 0xFF800000u, (size_t)P);  // ord(+inf)
        dim3 grid(P / 64, P / 64);
        ctx_gemm_pass<<<grid, 256, 0, stream>>>(Xn, Yn, C, P, rowMinU, rowZ, rowSmaxU, 0);
        fill_u32<<<(P + 255) / 256, 256, 0, stream>>>((unsigned int*)rowZ, 0u, (size_t)P);
        fill_u32<<<(P + 255) / 256, 256, 0, stream>>>(rowSmaxU, 0x007FFFFFu, (size_t)P);  // ord(-inf)
        ctx_gemm_pass<<<grid, 256, 0, stream>>>(Xn, Yn, C, P, rowMinU, rowZ, rowSmaxU, 1);
        fill_u32<<<1, 64, 0, stream>>>((unsigned int*)insideBuf, 0u, (size_t)1);
        ctx_row_accum<<<32, 256, 0, stream>>>(rowZ, rowSmaxU, insideBuf, P);
        ctx_finalize<<<1, 32, 0, stream>>>(insideBuf, (float*)d_out, WST[s], (float)C);
    }
}
